// PointFlow_down_6047313953097
// MI455X (gfx1250) — matrix operand order repaired and checked by reading, not compiled, not run on hardware
//
#include <hip/hip_runtime.h>

// ============================================================================
// PointFlow sparse down-conv pipeline for MI455X (gfx1250, wave32, WMMA).
//
//   h1  = relu(segsum(f16(feats) @ W1d[off1]) + b1d)
//   pf2 = h1 @ W11 + b11                      -> d_out[0 .. m1*64)
//   h2  = relu(segsum(f16(pf2) @ W2d[off2]) + b2d)
//   pf3 = h2 @ W22 + b22                      -> d_out[m1*64 .. (m1+m2)*64)
//
// All matmuls use V_WMMA_F32_16X16X32_F16 (f16 A/B, fp32 accumulation).
// The tap-dependent down-conv is made WMMA-dense by compacting point indices
// into 8 tap-uniform, 16-padded ranges, so every 16-row tile needs only
// 8 WMMAs with a single L0-resident weight matrix.  B fragments are hoisted
// into registers before the WMMA chain (no WMMA->VALU WAR hazards), and B
// addresses fold to base + immediate offsets.
// Segments (seg1/seg2) are sorted/contiguous by construction, so the
// segment-sum is a deterministic contiguous reduction (no FP atomics).
// ============================================================================

typedef __attribute__((ext_vector_type(16))) _Float16     v16h;
typedef __attribute__((ext_vector_type(8)))  float        v8f;
typedef __attribute__((ext_vector_type(8)))  unsigned int v8u;

union AFrag { v16h v; _Float16 h[16]; uint4 q[2]; };
union BFrag { v16h v; v8u u; };

#define CCH 64   // channels

static __device__ __forceinline__ v16h zero_v16h() {
    AFrag a;
#pragma unroll
    for (int i = 0; i < 16; ++i) a.h[i] = (_Float16)0.0f;
    return a.v;
}

static __device__ __forceinline__ v8f zero_v8f() {
    union { v8f v; float f[8]; } z;
#pragma unroll
    for (int i = 0; i < 8; ++i) z.f[i] = 0.0f;
    return z.v;
}

// ----------------------------------------------------------------------------
// Weight prepack: f32 row-major [K=64][N=64] -> f16 WMMA B-fragment layout.
// 18 matrices: [0..7]=W1d taps, [8]=W11, [9..16]=W2d taps, [17]=W22.
// Fragment (mat, t=ktile, nt=ntile): per lane 8 dwords, dword dw holds
// K = t*32 + (lane<16 ? 0 : 8) + (dw<4 ? 2*dw : 16+2*(dw-4)) (low half) and +1 (high half),
// N = nt*16 + (lane & 15).  Stored so each lane's 8 dwords are contiguous:
// dword index = ((mat*8 + t*4 + nt)*32 + lane)*8 + dw.
// ----------------------------------------------------------------------------
__global__ void prepack_weights(const float* __restrict__ W1d,
                                const float* __restrict__ W11,
                                const float* __restrict__ W2d,
                                const float* __restrict__ W22,
                                unsigned int* __restrict__ wfrag)
{
    int id = blockIdx.x * blockDim.x + threadIdx.x;   // 0 .. 18*8*32*8-1 = 36863
    int dw   = id & 7;
    int lane = (id >> 3) & 31;
    int frag = id >> 8;           // 0..143
    int mat  = frag >> 3;         // 0..17
    int sub  = frag & 7;
    int t    = sub >> 2;
    int nt   = sub & 3;

    const float* src;
    if      (mat < 8)  src = W1d + mat * 4096;
    else if (mat == 8) src = W11;
    else if (mat < 17) src = W2d + (mat - 9) * 4096;
    else               src = W22;

    int k0 = t * 32 + ((lane < 16) ? 0 : 8) + ((dw < 4) ? dw * 2 : 16 + (dw - 4) * 2);
    int nc = nt * 16 + (lane & 15);

    union { _Float16 h[2]; unsigned int u; } p;
    p.h[0] = (_Float16)src[(size_t)k0 * CCH + nc];
    p.h[1] = (_Float16)src[(size_t)(k0 + 1) * CCH + nc];
    wfrag[id] = p.u;
}

// ----------------------------------------------------------------------------
// Tap compaction (integer atomics only; output is order-invariant).
// meta layout (ints): [0..7]=cnt, [8..15]=fill, [16..24]=base_pad[0..8]
// ----------------------------------------------------------------------------
__global__ void init_compact(int* __restrict__ list, int* __restrict__ meta, int cap)
{
    int i = blockIdx.x * blockDim.x + threadIdx.x;
    if (i < cap) list[i] = -1;
    if (i < 8)   meta[i] = 0;           // cnt
}

__global__ void tap_hist(const int* __restrict__ off, int* __restrict__ meta, int n)
{
    int i = blockIdx.x * blockDim.x + threadIdx.x;
    if (i >= n) return;
    atomicAdd(&meta[off[i]], 1);
}

__global__ void tap_bases(int* __restrict__ meta)
{
    // single thread: base_pad[k] = 16-padded exclusive prefix of cnt; zero fill
    if (threadIdx.x == 0 && blockIdx.x == 0) {
        int acc = 0;
        meta[16] = 0;
        for (int k = 0; k < 8; ++k) {
            meta[8 + k] = 0;                        // fill[k] = 0
            acc += (meta[k] + 15) & ~15;            // round cnt up to 16
            meta[17 + k] = acc;                     // base_pad[k+1]
        }
    }
}

__global__ void tap_scatter(const int* __restrict__ off, int* __restrict__ meta,
                            int* __restrict__ list, int n)
{
    int i = blockIdx.x * blockDim.x + threadIdx.x;
    if (i >= n) return;
    int tap = off[i];
    int pos = meta[16 + tap] + atomicAdd(&meta[8 + tap], 1);
    list[pos] = i;
}

// ----------------------------------------------------------------------------
// Down-conv GEMM over tap-compacted index list:
//   msg[list[e], :] = f16(fin[list[e], :]) @ W[tap(e)]     (fp32 accumulate)
// One wave per 16-entry tile.  Every tile lies inside one tap range
// (ranges are 16-padded), so the tap -> weight lookup is wave-uniform and
// each tile needs only 2x4 = 8 WMMAs.  Sentinel entries (-1) get zero rows.
// ----------------------------------------------------------------------------
__global__ void down_gemm_gather(const float* __restrict__ fin,
                                 const int*   __restrict__ list,
                                 const int*   __restrict__ meta,
                                 const unsigned int* __restrict__ wfrag,  // 8 mats
                                 _Float16* __restrict__ msg)
{
    const int wave = threadIdx.x >> 5;
    const int lane = threadIdx.x & 31;
    const int t0   = (blockIdx.x * (blockDim.x >> 5) + wave) << 4;

    const int total = meta[16 + 8];            // base_pad[8] (uniform)
    if (t0 >= total) return;                   // wave-uniform exit

    int k = 0;                                 // tap of this tile (uniform)
#pragma unroll
    for (int j = 1; j < 8; ++j) k = (t0 >= meta[16 + j]) ? j : k;

    // All 8 B fragments of W[k] hoisted into registers (base + imm offsets).
    const unsigned int* wbase = wfrag + k * 2048 + lane * 8;
    v16h b[8];
#pragma unroll
    for (int f = 0; f < 8; ++f) {
        BFrag bf;
        bf.u = *(const v8u*)(wbase + f * 256);
        b[f] = bf.v;
    }

    const int idx   = list[t0 + (lane & 15)];  // source point of lane's A row
    const bool valid = (idx >= 0);
    const int rowc  = valid ? idx : 0;

    // A fragments (gathered row, f32 -> f16).  Lane<16: cols {c0..c0+7,
    // c0+16..c0+23}, c0 = t*32; lanes>=16 take the complementary +8 groups.
    const int cb = (lane < 16) ? 0 : 8;
    const float* fp = fin + (size_t)rowc * CCH;
    const v16h az = zero_v16h();
    v16h a[2];
#pragma unroll
    for (int t = 0; t < 2; ++t) {
        const int c0 = t * 32 + cb;
        float4 x0 = *(const float4*)(fp + c0);
        float4 x1 = *(const float4*)(fp + c0 + 4);
        float4 y0 = *(const float4*)(fp + c0 + 16);
        float4 y1 = *(const float4*)(fp + c0 + 20);
        AFrag af;
        af.h[0]  = (_Float16)x0.x; af.h[1]  = (_Float16)x0.y;
        af.h[2]  = (_Float16)x0.z; af.h[3]  = (_Float16)x0.w;
        af.h[4]  = (_Float16)x1.x; af.h[5]  = (_Float16)x1.y;
        af.h[6]  = (_Float16)x1.z; af.h[7]  = (_Float16)x1.w;
        af.h[8]  = (_Float16)y0.x; af.h[9]  = (_Float16)y0.y;
        af.h[10] = (_Float16)y0.z; af.h[11] = (_Float16)y0.w;
        af.h[12] = (_Float16)y1.x; af.h[13] = (_Float16)y1.y;
        af.h[14] = (_Float16)y1.z; af.h[15] = (_Float16)y1.w;
        a[t] = valid ? af.v : az;              // sentinel rows -> zero
    }

    v8f acc[4];
#pragma unroll
    for (int nt = 0; nt < 4; ++nt) acc[nt] = zero_v8f();

    // 8 back-to-back WMMAs; operands all resident in registers.
#pragma unroll
    for (int t = 0; t < 2; ++t) {
#pragma unroll
        for (int nt = 0; nt < 4; ++nt) {
            acc[nt] = __builtin_amdgcn_wmma_f32_16x16x32_f16(
                false, a[t], false, b[t * 4 + nt], (short)0, acc[nt], false, false);
        }
    }

    // C/D layout: VGPR d -> (M = d or d+8, N = lane&15 within ntile).
    int rsel[8];
#pragma unroll
    for (int d = 0; d < 8; ++d)
        rsel[d] = list[t0 + ((lane < 16) ? d : d + 8)];

#pragma unroll
    for (int nt = 0; nt < 4; ++nt) {
#pragma unroll
        for (int d = 0; d < 8; ++d) {
            if (rsel[d] >= 0)
                msg[(size_t)rsel[d] * CCH + nt * 16 + (lane & 15)] =
                    (_Float16)acc[nt][d];
        }
    }
}

// ----------------------------------------------------------------------------
// starts[j] = first input index of segment j (seg is sorted, every segment
// non-empty), starts[m] = n.
// ----------------------------------------------------------------------------
__global__ void build_starts(const int* __restrict__ seg,
                             int* __restrict__ starts, int n, int m)
{
    int i = blockIdx.x * blockDim.x + threadIdx.x;
    if (i >= n) return;
    int s = seg[i];
    if (i == 0) { starts[0] = 0; starts[m] = n; }
    else if (s != seg[i - 1]) starts[s] = i;
}

// ----------------------------------------------------------------------------
// h[j,c] = relu(bias[c] + sum_{i in segment j} msg[i,c])   (fp32 accumulate)
// ----------------------------------------------------------------------------
__global__ void segsum_relu(const _Float16* __restrict__ msg,
                            const int*   __restrict__ starts,
                            const float* __restrict__ bias,
                            _Float16* __restrict__ h, int m)
{
    int idx = blockIdx.x * blockDim.x + threadIdx.x;
    if (idx >= m * CCH) return;
    int j = idx >> 6;
    int c = idx & 63;
    float s = bias[c];
    int e = starts[j + 1];
    for (int i = starts[j]; i < e; ++i)
        s += (float)msg[(size_t)i * CCH + c];
    h[idx] = (_Float16)fmaxf(s, 0.0f);
}

// ----------------------------------------------------------------------------
// 1x1 conv: out[i,:] = hin[i,:] @ W + b   (f16 A/B, fp32 accum, f32 out)
// ----------------------------------------------------------------------------
__global__ void onexone_gemm(const _Float16* __restrict__ hin,
                             const unsigned int* __restrict__ wfrag,  // 1 mat
                             const float* __restrict__ bias,
                             float* __restrict__ out, int n)
{
    const int wave  = threadIdx.x >> 5;
    const int lane  = threadIdx.x & 31;
    const int tiles = (n + 15) >> 4;
    const int tile  = blockIdx.x * (blockDim.x >> 5) + wave;
    if (tile >= tiles) return;

    const int row0  = tile << 4;
    const int myrow = row0 + (lane & 15);
    const int rowc  = (myrow < n) ? myrow : (n - 1);
    const int cb    = (lane < 16) ? 0 : 8;
    const _Float16* fp = hin + (size_t)rowc * CCH;

    // All 8 B fragments hoisted (base + imm offsets).
    const unsigned int* wbase = wfrag + lane * 8;
    v16h b[8];
#pragma unroll
    for (int f = 0; f < 8; ++f) {
        BFrag bf;
        bf.u = *(const v8u*)(wbase + f * 256);
        b[f] = bf.v;
    }

    v16h a[2];
#pragma unroll
    for (int t = 0; t < 2; ++t) {
        AFrag af;
        af.q[0] = *(const uint4*)(fp + t * 32 + cb);
        af.q[1] = *(const uint4*)(fp + t * 32 + cb + 16);
        a[t] = af.v;
    }

    v8f acc[4];
#pragma unroll
    for (int nt = 0; nt < 4; ++nt) acc[nt] = zero_v8f();

#pragma unroll
    for (int t = 0; t < 2; ++t) {
#pragma unroll
        for (int nt = 0; nt < 4; ++nt) {
            acc[nt] = __builtin_amdgcn_wmma_f32_16x16x32_f16(
                false, a[t], false, b[t * 4 + nt], (short)0, acc[nt], false, false);
        }
    }

    float bv[4];
#pragma unroll
    for (int nt = 0; nt < 4; ++nt) bv[nt] = bias[nt * 16 + (lane & 15)];

#pragma unroll
    for (int nt = 0; nt < 4; ++nt) {
#pragma unroll
        for (int d = 0; d < 8; ++d) {
            int r = row0 + ((lane < 16) ? d : d + 8);
            if (r < n)
                out[(size_t)r * CCH + nt * 16 + (lane & 15)] = acc[nt][d] + bv[nt];
        }
    }
}

// ============================================================================
// Host side
// ============================================================================
extern "C" void kernel_launch(void* const* d_in, const int* in_sizes, int n_in,
                              void* d_out, int out_size, void* d_ws, size_t ws_size,
                              hipStream_t stream)
{
    const float* feats = (const float*)d_in[0];
    const float* W1d   = (const float*)d_in[1];
    const float* b1d   = (const float*)d_in[2];
    const float* W11   = (const float*)d_in[3];
    const float* b11   = (const float*)d_in[4];
    const float* W2d   = (const float*)d_in[5];
    const float* b2d   = (const float*)d_in[6];
    const float* W22   = (const float*)d_in[7];
    const float* b22   = (const float*)d_in[8];
    const int*   seg1  = (const int*)d_in[9];
    const int*   off1  = (const int*)d_in[10];
    const int*   seg2  = (const int*)d_in[11];
    const int*   off2  = (const int*)d_in[12];
    // d_in[13]/d_in[14] (m1,m2 device scalars) unused: sizes recovered on host.

    const int n  = in_sizes[9];              // N   (len(seg1))
    const int m1 = in_sizes[11];             // m1  (len(seg2))
    const int m2 = out_size / CCH - m1;      // m2

    // Workspace layout (all regions fully rewritten every call):
    //   [0MB,  1MB)     wfrag (18 mats * 8KB f16 fragments = 144KB)
    //   [1MB,  2MB)     meta  (cnt[8], fill[8], base_pad[9])
    //   [2MB,  6MB)     starts1 (m1+1 ints)
    //   [6MB, 10MB)     starts2 (m2+1 ints)
    //   [10MB,16MB)     list   (tap-compacted indices, up to n+128 ints)
    //   [16MB, +128MB)  msgBuf (f16, max(n,m1)*64)
    //   [16MB+128MB,..) hBuf   (f16, max(m1,m2)*64)
    char* ws = (char*)d_ws;
    const size_t MB = 1u << 20;
    unsigned int* wfrag = (unsigned int*)ws;
    int*       meta     = (int*)(ws + 1 * MB);
    int*       starts1  = (int*)(ws + 2 * MB);
    int*       starts2  = (int*)(ws + 6 * MB);
    int*       list     = (int*)(ws + 10 * MB);
    _Float16*  msgBuf   = (_Float16*)(ws + 16 * MB);
    _Float16*  hBuf     = (_Float16*)(ws + 16 * MB + ((size_t)1 << 27));

    float* pf2 = (float*)d_out;
    float* pf3 = (float*)d_out + (size_t)m1 * CCH;

    const dim3 blk(256);                     // 8 waves per block
    auto gemm_blocks = [](int rows) -> unsigned int {
        int tiles = (rows + 15) / 16;
        return (unsigned int)((tiles + 7) / 8);
    };

    prepack_weights<<<144, blk, 0, stream>>>(W1d, W11, W2d, W22, wfrag);

    // ---- stage 1: pf2 = (relu(downconv(feats))) @ W11 + b11 ----
    {
        const int cap = n + 128;             // 16-padding of 8 tap ranges
        init_compact<<<(cap + 255) / 256, blk, 0, stream>>>(list, meta, cap);
        tap_hist<<<(n + 255) / 256, blk, 0, stream>>>(off1, meta, n);
        tap_bases<<<1, 1, 0, stream>>>(meta);
        tap_scatter<<<(n + 255) / 256, blk, 0, stream>>>(off1, meta, list, n);
        down_gemm_gather<<<gemm_blocks(cap), blk, 0, stream>>>(
            feats, list, meta, wfrag, msgBuf);
    }
    build_starts<<<(n + 255) / 256, blk, 0, stream>>>(seg1, starts1, n, m1);
    segsum_relu<<<((unsigned int)m1 * CCH + 255) / 256, blk, 0, stream>>>(
        msgBuf, starts1, b1d, hBuf, m1);
    onexone_gemm<<<gemm_blocks(m1), blk, 0, stream>>>(
        hBuf, wfrag + 8 * 2048, b11, pf2, m1);

    // ---- stage 2: pf3 = (relu(downconv(pf2))) @ W22 + b22 ----
    {
        const int cap = m1 + 128;
        init_compact<<<(cap + 255) / 256, blk, 0, stream>>>(list, meta, cap);
        tap_hist<<<(m1 + 255) / 256, blk, 0, stream>>>(off2, meta, m1);
        tap_bases<<<1, 1, 0, stream>>>(meta);
        tap_scatter<<<(m1 + 255) / 256, blk, 0, stream>>>(off2, meta, list, m1);
        down_gemm_gather<<<gemm_blocks(cap), blk, 0, stream>>>(
            pf2, list, meta, wfrag + 9 * 2048, msgBuf);
    }
    build_starts<<<(m1 + 255) / 256, blk, 0, stream>>>(seg2, starts2, m1, m2);
    segsum_relu<<<((unsigned int)m2 * CCH + 255) / 256, blk, 0, stream>>>(
        msgBuf, starts2, b2d, hBuf, m2);
    onexone_gemm<<<gemm_blocks(m2), blk, 0, stream>>>(
        hBuf, wfrag + 17 * 2048, b22, pf3, m2);
}
